// MemoryRetrieval_23313082483185
// MI455X (gfx1250) — matrix operand order repaired and checked by reading, not compiled, not run on hardware
//
#include <hip/hip_runtime.h>

// ---------------------------------------------------------------------------
// Types for CDNA5 WMMA (wave32, 16x16x32 f16 -> f32)
// ---------------------------------------------------------------------------
typedef __attribute__((ext_vector_type(16))) _Float16 v16h;
typedef __attribute__((ext_vector_type(8)))  _Float16 v8h;
typedef __attribute__((ext_vector_type(8)))  float    v8f;

#define DEV __device__ __forceinline__

constexpr int   TSTEPS = 4;
constexpr int   NB     = 16;     // batch
constexpr int   DIMC   = 384;
constexpr int   NHEAD  = 8;
constexpr int   CHD    = 48;     // DIMC / NHEAD
constexpr int   EXPC   = 1536;
constexpr int   DHD    = 192;    // EXPC / NHEAD
constexpr int   NSP    = 576;    // 24*24 tokens
constexpr float THRESH = 0.5f;
constexpr float DECAY  = 0.25f;
constexpr float SCALE2 = 0.28867513459481287f;  // (1/sqrt(48)) * 2

// ---------------------------------------------------------------------------
// WMMA fragment helpers (layouts per CDNA5 ISA 7.12.2)
// A 16x32 f16: lane l -> row m=l&15, half=l>>4; K runs [half*8..+8) and [16+half*8..+8)
// B 32x16 f16: lane l -> col n=l&15, half=l>>4; K runs [half*8..+8) and [16+half*8..+8)
// C 16x16 f32: lane l -> col n=l&15; VGPR r -> row m = r + (l>>4)*8
// ---------------------------------------------------------------------------
DEV v16h load_a_frag(const _Float16* __restrict__ p, int ld) {
  const int l = threadIdx.x & 31;
  const _Float16* row = p + (size_t)(l & 15) * ld + (l >> 4) * 8;
  v8h lo = *(const v8h*)(row);
  v8h hi = *(const v8h*)(row + 16);
  v16h r;
#pragma unroll
  for (int i = 0; i < 8; ++i) { r[i] = lo[i]; r[i + 8] = hi[i]; }
  return r;
}

DEV v16h load_b_frag(const _Float16* __restrict__ p, int ld) {
  const int l = threadIdx.x & 31;
  const _Float16* col = p + (size_t)(l & 15) * ld + (l >> 4) * 8;
  v8h lo = *(const v8h*)(col);
  v8h hi = *(const v8h*)(col + 16);
  v16h r;
#pragma unroll
  for (int i = 0; i < 8; ++i) { r[i] = lo[i]; r[i + 8] = hi[i]; }
  return r;
}

DEV v8f wmma_f16(v16h a, v16h b, v8f c) {
  return __builtin_amdgcn_wmma_f32_16x16x32_f16(false, a, false, b, (short)0, c,
                                                false, false);
}

// ---------------------------------------------------------------------------
// Generic spike GEMM with fused temporal-LIF epilogue.
//   Y_t[m,o] = sum_k A_t[m,k] * W_t[o,k];  v = Y*scale[o]+bias[o] (+shortcut)
//   MODE 0: write float v (t=0 only)
//   MODE 1: write fp16 spike (v>thresh), single step
//   MODE 2: LIF across TT timesteps -> fp16 spikes (normal and/or transposed)
//   MODE 3: write float v, then 4-step constant-input LIF -> fp16 spikes
// Block: 256 thr = 8 waves (4M x 2N); block tile 64M x 64N; wave 16M x 32N x TT.
// ---------------------------------------------------------------------------
template<int TT, int MODE, bool PER_T_W, bool ADD_SC>
__global__ __launch_bounds__(256) void gemm_spike(
    const _Float16* __restrict__ A, size_t aT,
    const _Float16* __restrict__ W, size_t wT,
    const float* __restrict__ sc, const float* __restrict__ bi,
    const float* __restrict__ shortcut,
    _Float16* __restrict__ outN, size_t oN_t,
    _Float16* __restrict__ outT, size_t oT_t, size_t oT_b,
    float* __restrict__ outF,
    int K, int Nout)
{
  const int lane = threadIdx.x & 31;
  const int w    = threadIdx.x >> 5;
  const int m0   = blockIdx.y * 64 + (w >> 1) * 16;
  const int n0   = blockIdx.x * 64 + (w & 1) * 32;

  v8f acc[TT][2] = {};

  for (int k0 = 0; k0 < K; k0 += 32) {
    v16h b0 = {}, b1 = {};
    if constexpr (!PER_T_W) {
      b0 = load_b_frag(W + (size_t)n0 * K + k0, K);
      b1 = load_b_frag(W + (size_t)(n0 + 16) * K + k0, K);
      if (k0 + 32 < K)
        __builtin_prefetch(W + (size_t)n0 * K + k0 + 32, 0, 3);
    }
#pragma unroll
    for (int t = 0; t < TT; ++t) {
      if constexpr (PER_T_W) {
        const _Float16* Wt = W + (size_t)t * wT;
        b0 = load_b_frag(Wt + (size_t)n0 * K + k0, K);
        b1 = load_b_frag(Wt + (size_t)(n0 + 16) * K + k0, K);
      }
      v16h a = load_a_frag(A + (size_t)t * aT + (size_t)m0 * K + k0, K);
      acc[t][0] = wmma_f16(a, b0, acc[t][0]);
      acc[t][1] = wmma_f16(a, b1, acc[t][1]);
    }
  }

  const int lm = lane & 15, lh = lane >> 4;
#pragma unroll
  for (int j = 0; j < 2; ++j) {
    const int o = n0 + j * 16 + lm;
    const float scl = sc[o], bb = bi[o];
#pragma unroll
    for (int r = 0; r < 8; ++r) {
      const int m = m0 + r + lh * 8;
      if constexpr (MODE == 0) {
        outF[(size_t)m * Nout + o] = acc[0][j][r] * scl + bb;
      } else if constexpr (MODE == 1) {
        float v = acc[0][j][r] * scl + bb;
        outN[(size_t)m * Nout + o] = (_Float16)(v > THRESH ? 1.f : 0.f);
      } else if constexpr (MODE == 3) {
        float v = acc[0][j][r] * scl + bb;
        outF[(size_t)m * Nout + o] = v;
        float mem = 0.f, sp = 0.f;
#pragma unroll
        for (int t = 0; t < 4; ++t) {
          mem = mem * DECAY * (1.f - sp) + v;
          sp  = mem > THRESH ? 1.f : 0.f;
          outN[(size_t)t * oN_t + (size_t)m * Nout + o] = (_Float16)sp;
        }
      } else {  // MODE 2: LIF across TT
        float mem = 0.f, sp = 0.f;
#pragma unroll
        for (int t = 0; t < TT; ++t) {
          float scl_t = scl, bb_t = bb;
          if constexpr (PER_T_W) {
            scl_t = sc[(size_t)t * Nout + o];
            bb_t  = bi[(size_t)t * Nout + o];
          }
          float v = acc[t][j][r] * scl_t + bb_t;
          if constexpr (ADD_SC) v += shortcut[(size_t)m * Nout + o];
          mem = mem * DECAY * (1.f - sp) + v;
          sp  = mem > THRESH ? 1.f : 0.f;
          _Float16 hsp = (_Float16)sp;
          if (outN)
            outN[(size_t)t * oN_t + (size_t)m * Nout + o] = hsp;
          if (outT)
            outT[(size_t)t * oT_t + (size_t)(m / NSP) * oT_b +
                 (size_t)o * NSP + (m % NSP)] = hsp;
        }
      }
    }
  }
}

// ---------------------------------------------------------------------------
// kv einsum per (t,b,h): kv[c(48), d(192)] = sum_n kT[c,n] * vT[d,n]
// Output transposed + K-padded: kvT[t,b,h, d, 64] fp16 (cols 48..63 pre-zeroed),
// with attention scale (2/sqrt(48)) folded in.
// Block 128 thr = 4 waves; wave: 48M x 16N; grid (3, 1, t*b*h=512).
// ---------------------------------------------------------------------------
__global__ __launch_bounds__(128) void kv_kernel(
    const _Float16* __restrict__ kT, const _Float16* __restrict__ vT,
    _Float16* __restrict__ kvT)
{
  const int lane  = threadIdx.x & 31;
  const int w     = threadIdx.x >> 5;
  const int ncol0 = blockIdx.x * 64 + w * 16;
  const int tbh   = blockIdx.z;
  const int t = tbh >> 7, b = (tbh >> 3) & 15, h = tbh & 7;

  const _Float16* kb = kT + (size_t)((t * NB + b) * DIMC + h * CHD) * NSP;
  const _Float16* vb = vT + (size_t)((t * NB + b) * EXPC + h * DHD) * NSP;

  v8f acc[3] = {};
  for (int k0 = 0; k0 < NSP; k0 += 32) {
    v16h bf = load_b_frag(vb + (size_t)ncol0 * NSP + k0, NSP);
#pragma unroll
    for (int mt = 0; mt < 3; ++mt) {
      v16h a = load_a_frag(kb + (size_t)(mt * 16) * NSP + k0, NSP);
      acc[mt] = wmma_f16(a, bf, acc[mt]);
    }
  }

  _Float16* ob = kvT + (size_t)((t * NB + b) * NHEAD + h) * DHD * 64;
  const int lm = lane & 15, lh = lane >> 4;
#pragma unroll
  for (int mt = 0; mt < 3; ++mt)
#pragma unroll
    for (int r = 0; r < 8; ++r) {
      const int c = mt * 16 + r + lh * 8;
      const int d = ncol0 + lm;
      ob[(size_t)d * 64 + c] = (_Float16)(acc[mt][r] * SCALE2);
    }
}

// ---------------------------------------------------------------------------
// attn_out per (b,h), T-fused: x_t[n,d] = sum_c q_t[n,c+h*48] * kvT_t[d,c]
// K = 64 (48 real + 16 zero-padded in kvT). LIF across t -> spikes.
// EPI==2 additionally accumulates spike sums over n into pooled[t,b,d]
// (cross-lane shfl + global float atomics).
// Block 256 thr = 8 waves (4M x 2N); grid (6, 9, b*h=128).
// ---------------------------------------------------------------------------
template<int EPI>
__global__ __launch_bounds__(256) void attn_kernel(
    const _Float16* __restrict__ Q,    // [t,b,n,384] spikes
    const _Float16* __restrict__ kvT,  // [t,b,h,192,64] fp16, scale folded
    _Float16* __restrict__ outSp,      // [t,b,n,1536] spikes
    float* __restrict__ pooled)        // [t,b,1536] spike sums (pre-zeroed)
{
  const int lane = threadIdx.x & 31;
  const int w    = threadIdx.x >> 5;
  const int m0   = blockIdx.y * 64 + (w >> 1) * 16;   // token rows
  const int n0   = blockIdx.x * 32 + (w & 1) * 16;    // d cols within head
  const int bh   = blockIdx.z;
  const int b = bh >> 3, h = bh & 7;

  const size_t qT   = (size_t)NB * NSP * DIMC;
  const size_t kvTt = (size_t)NB * NHEAD * DHD * 64;
  const _Float16* qb = Q + (size_t)(b * NSP) * DIMC + h * CHD;
  const _Float16* kb = kvT + (size_t)(b * NHEAD + h) * DHD * 64;

  v8f acc[4] = {};
#pragma unroll
  for (int k0 = 0; k0 < 64; k0 += 32)
#pragma unroll
    for (int t = 0; t < 4; ++t) {
      v16h a  = load_a_frag(qb + (size_t)t * qT + (size_t)m0 * DIMC + k0, DIMC);
      v16h bf = load_b_frag(kb + (size_t)t * kvTt + (size_t)n0 * 64 + k0, 64);
      acc[t] = wmma_f16(a, bf, acc[t]);
    }

  const int lm = lane & 15, lh = lane >> 4;
  const int dabs = h * DHD + n0 + lm;
  float psum[4] = {0.f, 0.f, 0.f, 0.f};
#pragma unroll
  for (int r = 0; r < 8; ++r) {
    const int m = m0 + r + lh * 8;
    float mem = 0.f, sp = 0.f;
#pragma unroll
    for (int t = 0; t < 4; ++t) {
      mem = mem * DECAY * (1.f - sp) + acc[t][r];
      sp  = mem > THRESH ? 1.f : 0.f;
      outSp[((size_t)(t * NB + b) * NSP + m) * EXPC + dabs] = (_Float16)sp;
      if constexpr (EPI == 2) psum[t] += sp;
    }
  }
  if constexpr (EPI == 2) {
#pragma unroll
    for (int t = 0; t < 4; ++t) {
      float v = psum[t] + __shfl_xor(psum[t], 16, 32);
      if (lane < 16)
        atomicAdd(&pooled[(size_t)(t * NB + b) * EXPC + dabs], v);
    }
  }
}

// ---------------------------------------------------------------------------
// Elementwise / small kernels
// ---------------------------------------------------------------------------
__global__ void f2h_kernel(const float* __restrict__ s, _Float16* __restrict__ d,
                           size_t n) {
  size_t i = (size_t)blockIdx.x * blockDim.x + threadIdx.x;
  if (i < n) d[i] = (_Float16)s[i];
}

// templates [4,16,384,576] + t_pos -> LIF -> spikes [t,b,n,c]
__global__ void lif_pos_tm_kernel(const float* __restrict__ tmpl,
                                  const float* __restrict__ tpos,
                                  _Float16* __restrict__ tm_sp) {
  size_t idx = (size_t)blockIdx.x * blockDim.x + threadIdx.x;
  if (idx >= (size_t)NB * DIMC * NSP) return;
  const int n = idx % NSP;
  const size_t bc = idx / NSP;
  const int b = bc / DIMC, c = bc % DIMC;
  float mem = 0.f, sp = 0.f;
  for (int t = 0; t < TSTEPS; ++t) {
    float x = tmpl[((size_t)(t * NB + b) * DIMC + c) * NSP + n] +
              tpos[((size_t)t * DIMC + c) * NSP + n];
    mem = mem * DECAY * (1.f - sp) + x;
    sp  = mem > THRESH ? 1.f : 0.f;
    tm_sp[((size_t)(t * NB + b) * NSP + n) * DIMC + c] = (_Float16)sp;
  }
}

// search + s_pos -> single-step spike [b,n,c]
__global__ void lif_pos_sr_kernel(const float* __restrict__ srch,
                                  const float* __restrict__ spos,
                                  _Float16* __restrict__ sr_sp) {
  size_t idx = (size_t)blockIdx.x * blockDim.x + threadIdx.x;
  if (idx >= (size_t)NB * DIMC * NSP) return;
  const int n = idx % NSP;
  const size_t bc = idx / NSP;
  const int b = bc / DIMC, c = bc % DIMC;
  float x = srch[((size_t)b * DIMC + c) * NSP + n] +
            spos[(size_t)c * NSP + n];
  sr_sp[((size_t)b * NSP + n) * DIMC + c] =
      (_Float16)(x > THRESH ? 1.f : 0.f);
}

// depthwise 3x3 + BN per timestep (per-t weights) + LIF across t
__global__ void dw_lif_kernel(const _Float16* __restrict__ in,
                              const float* __restrict__ wdw,
                              const float* __restrict__ sdw,
                              const float* __restrict__ bdw,
                              _Float16* __restrict__ out) {
  size_t idx = (size_t)blockIdx.x * blockDim.x + threadIdx.x;
  if (idx >= (size_t)NB * NSP * EXPC) return;
  const int c = idx % EXPC;
  const int n = (idx / EXPC) % NSP;
  const int b = idx / ((size_t)EXPC * NSP);
  const int y = n / 24, x = n % 24;
  float mem = 0.f, sp = 0.f;
  for (int t = 0; t < TSTEPS; ++t) {
    float a = 0.f;
    const float* wt = wdw + ((size_t)t * EXPC + c) * 9;
    for (int dy = -1; dy <= 1; ++dy) {
      const int yy = y + dy;
      if (yy < 0 || yy >= 24) continue;
      for (int dx = -1; dx <= 1; ++dx) {
        const int xx = x + dx;
        if (xx < 0 || xx >= 24) continue;
        a += (float)in[((size_t)(t * NB + b) * NSP + yy * 24 + xx) * EXPC + c] *
             wt[(dy + 1) * 3 + (dx + 1)];
      }
    }
    float v = a * sdw[t * EXPC + c] + bdw[t * EXPC + c];
    mem = mem * DECAY * (1.f - sp) + v;
    sp  = mem > THRESH ? 1.f : 0.f;
    out[((size_t)(t * NB + b) * NSP + n) * EXPC + c] = (_Float16)sp;
  }
}

// gate logits: gl[t,b,d] = (pooled[t,b,:]/576) @ wgate[:,d]
__global__ void gate_logits_kernel(const float* __restrict__ pooled,
                                   const float* __restrict__ wgate,
                                   float* __restrict__ gl) {
  const int idx = blockIdx.x * blockDim.x + threadIdx.x;
  if (idx >= TSTEPS * NB * EXPC) return;
  const int d = idx % EXPC;
  const int tb = idx / EXPC;
  const float* pr = pooled + (size_t)tb * EXPC;
  float s = 0.f;
  for (int c = 0; c < EXPC; ++c) s += pr[c] * wgate[(size_t)c * EXPC + d];
  gl[idx] = s * (1.f / (float)NSP);
}

// softmax over t (axis 0), in place
__global__ void softmax_t_kernel(float* __restrict__ gl) {
  const int idx = blockIdx.x * blockDim.x + threadIdx.x;
  if (idx >= NB * EXPC) return;
  float v[TSTEPS], mx = -1e30f;
  for (int t = 0; t < TSTEPS; ++t) {
    v[t] = gl[(size_t)t * NB * EXPC + idx];
    mx = fmaxf(mx, v[t]);
  }
  float s = 0.f;
  for (int t = 0; t < TSTEPS; ++t) { v[t] = __expf(v[t] - mx); s += v[t]; }
  const float inv = 1.f / s;
  for (int t = 0; t < TSTEPS; ++t) gl[(size_t)t * NB * EXPC + idx] = v[t] * inv;
}

// gated temporal sum + single-step spike -> [b,n,1536]
__global__ void gate_combine_kernel(const _Float16* __restrict__ xs,
                                    const float* __restrict__ gw,
                                    _Float16* __restrict__ gsp) {
  size_t idx = (size_t)blockIdx.x * blockDim.x + threadIdx.x;
  if (idx >= (size_t)NB * NSP * EXPC) return;
  const int d = idx % EXPC;
  const size_t bn = idx / EXPC;
  const int b = bn / NSP, n = bn % NSP;
  float s = 0.f;
  for (int t = 0; t < TSTEPS; ++t)
    s += (float)xs[((size_t)(t * NB + b) * NSP + n) * EXPC + d] *
         gw[(size_t)t * NB * EXPC + (size_t)b * EXPC + d];
  gsp[idx] = (_Float16)(s > THRESH ? 1.f : 0.f);
}

// s2 = search + retr; also single-step spike of s2 for MLP
__global__ void s2_kernel(const float* __restrict__ srch,
                          const float* __restrict__ retr,
                          float* __restrict__ s2f,
                          _Float16* __restrict__ s2sp) {
  size_t idx = (size_t)blockIdx.x * blockDim.x + threadIdx.x;
  if (idx >= (size_t)NB * NSP * DIMC) return;
  const int c = idx % DIMC;
  const size_t bn = idx / DIMC;
  const int b = bn / NSP, n = bn % NSP;
  float v = srch[((size_t)b * DIMC + c) * NSP + n] + retr[idx];
  s2f[idx]  = v;
  s2sp[idx] = (_Float16)(v > THRESH ? 1.f : 0.f);
}

// out[b,c,n] = s2[b,n,c] + y[b,n,c]  (transpose back to channel-major)
__global__ void final_kernel(const float* __restrict__ s2f,
                             const float* __restrict__ yf,
                             float* __restrict__ out) {
  size_t idx = (size_t)blockIdx.x * blockDim.x + threadIdx.x;
  if (idx >= (size_t)NB * DIMC * NSP) return;
  const int n = idx % NSP;
  const size_t bc = idx / NSP;
  const int b = bc / DIMC, c = bc % DIMC;
  const size_t src = ((size_t)b * NSP + n) * DIMC + c;
  out[idx] = s2f[src] + yf[src];
}

// ---------------------------------------------------------------------------
// Host orchestration
// ---------------------------------------------------------------------------
extern "C" void kernel_launch(void* const* d_in, const int* in_sizes, int n_in,
                              void* d_out, int out_size, void* d_ws, size_t ws_size,
                              hipStream_t stream) {
  (void)in_sizes; (void)n_in; (void)out_size; (void)ws_size;
  const float* templates = (const float*)d_in[0];
  const float* search    = (const float*)d_in[1];
  const float* s_pos  = (const float*)d_in[2];
  const float* t_pos  = (const float*)d_in[3];
  const float* wq = (const float*)d_in[4];
  const float* sq = (const float*)d_in[5];
  const float* bq = (const float*)d_in[6];
  const float* wk = (const float*)d_in[7];
  const float* sk = (const float*)d_in[8];
  const float* bk = (const float*)d_in[9];
  const float* wv = (const float*)d_in[10];
  const float* sv = (const float*)d_in[11];
  const float* bv = (const float*)d_in[12];
  const float* wproj = (const float*)d_in[13];
  const float* sproj = (const float*)d_in[14];
  const float* bproj = (const float*)d_in[15];
  const float* wgate = (const float*)d_in[16];
  const float* wdw = (const float*)d_in[17];
  const float* sdw = (const float*)d_in[18];
  const float* bdw = (const float*)d_in[19];
  const float* wpw = (const float*)d_in[20];
  const float* spw = (const float*)d_in[21];
  const float* bpw = (const float*)d_in[22];
  const float* wfuse = (const float*)d_in[23];
  const float* sfuse = (const float*)d_in[24];
  const float* bfuse = (const float*)d_in[25];
  const float* wfc1 = (const float*)d_in[26];
  const float* sfc1 = (const float*)d_in[27];
  const float* bfc1 = (const float*)d_in[28];
  const float* wfc2 = (const float*)d_in[29];
  const float* sfc2 = (const float*)d_in[30];
  const float* bfc2 = (const float*)d_in[31];

  // bump allocator over workspace (256B aligned, 256B slack per buffer)
  char* wsp = (char*)d_ws;
  auto alloc = [&](size_t bytes) -> void* {
    void* r = (void*)wsp;
    wsp += ((bytes + 255) & ~(size_t)255) + 256;
    return r;
  };

  constexpr size_t M = (size_t)NB * NSP;  // 9216 (grid.y = 144)
  const size_t TBNC = (size_t)TSTEPS * NB * NSP * DIMC;
  const size_t TBNE = (size_t)TSTEPS * NB * NSP * EXPC;

  _Float16* wq16   = (_Float16*)alloc(sizeof(_Float16) * DIMC * DIMC);
  _Float16* wk16   = (_Float16*)alloc(sizeof(_Float16) * DIMC * DIMC);
  _Float16* wv16   = (_Float16*)alloc(sizeof(_Float16) * EXPC * DIMC);
  _Float16* wpw16  = (_Float16*)alloc(sizeof(_Float16) * TSTEPS * DIMC * EXPC);
  _Float16* wfu16  = (_Float16*)alloc(sizeof(_Float16) * DIMC * DIMC);
  _Float16* wpj16  = (_Float16*)alloc(sizeof(_Float16) * DIMC * EXPC);
  _Float16* wf116  = (_Float16*)alloc(sizeof(_Float16) * EXPC * DIMC);
  _Float16* wf216  = (_Float16*)alloc(sizeof(_Float16) * DIMC * EXPC);

  _Float16* tm_sp  = (_Float16*)alloc(sizeof(_Float16) * TBNC);
  _Float16* sr_sp  = (_Float16*)alloc(sizeof(_Float16) * M * DIMC);
  _Float16* k_spT  = (_Float16*)alloc(sizeof(_Float16) * TBNC);
  _Float16* v_spT  = (_Float16*)alloc(sizeof(_Float16) * TBNE);
  _Float16* q_sp   = (_Float16*)alloc(sizeof(_Float16) * TBNC);
  float*    qsf    = (float*)   alloc(sizeof(float)    * M * DIMC);
  _Float16* kvT    = (_Float16*)alloc(sizeof(_Float16) * (size_t)TSTEPS * NB * NHEAD * DHD * 64);
  _Float16* qq_sp  = (_Float16*)alloc(sizeof(_Float16) * TBNE);
  _Float16* dw_sp  = (_Float16*)alloc(sizeof(_Float16) * TBNE);
  _Float16* pw_sp  = (_Float16*)alloc(sizeof(_Float16) * TBNC);
  _Float16* fu_sp  = (_Float16*)alloc(sizeof(_Float16) * TBNC);
  _Float16* x_sp   = (_Float16*)alloc(sizeof(_Float16) * TBNE);
  float*    pooled = (float*)   alloc(sizeof(float)    * TSTEPS * NB * EXPC);
  float*    gl     = (float*)   alloc(sizeof(float)    * TSTEPS * NB * EXPC);
  _Float16* gatesp = (_Float16*)alloc(sizeof(_Float16) * M * EXPC);
  float*    retr   = (float*)   alloc(sizeof(float)    * M * DIMC);
  float*    s2f    = (float*)   alloc(sizeof(float)    * M * DIMC);
  _Float16* s2sp   = (_Float16*)alloc(sizeof(_Float16) * M * DIMC);
  _Float16* h_sp   = (_Float16*)alloc(sizeof(_Float16) * M * EXPC);
  float*    yf     = (float*)   alloc(sizeof(float)    * M * DIMC);

  auto f2h = [&](const float* s, _Float16* d, size_t n) {
    f2h_kernel<<<(unsigned)((n + 255) / 256), 256, 0, stream>>>(s, d, n);
  };
  f2h(wq, wq16, (size_t)DIMC * DIMC);
  f2h(wk, wk16, (size_t)DIMC * DIMC);
  f2h(wv, wv16, (size_t)EXPC * DIMC);
  f2h(wpw, wpw16, (size_t)TSTEPS * DIMC * EXPC);
  f2h(wfuse, wfu16, (size_t)DIMC * DIMC);
  f2h(wproj, wpj16, (size_t)DIMC * EXPC);
  f2h(wfc1, wf116, (size_t)EXPC * DIMC);
  f2h(wfc2, wf216, (size_t)DIMC * EXPC);

  hipMemsetAsync(kvT, 0, sizeof(_Float16) * (size_t)TSTEPS * NB * NHEAD * DHD * 64, stream);
  hipMemsetAsync(pooled, 0, sizeof(float) * TSTEPS * NB * EXPC, stream);

  const unsigned gE = (unsigned)((M * DIMC + 255) / 256);          // 13824
  lif_pos_tm_kernel<<<gE, 256, 0, stream>>>(templates, t_pos, tm_sp);
  lif_pos_sr_kernel<<<gE, 256, 0, stream>>>(search, s_pos, sr_sp);

  const dim3 blk(256);
  const size_t aT_c = (size_t)NB * NSP * DIMC;   // per-t stride, 384-ch spikes
  const size_t aT_e = (size_t)NB * NSP * EXPC;   // per-t stride, 1536-ch spikes

  // k: tm_sp @ wk -> LIF -> k_spT [t,b,c,n]
  gemm_spike<4, 2, false, false><<<dim3(6, 144), blk, 0, stream>>>(
      tm_sp, aT_c, wk16, 0, sk, bk, nullptr,
      nullptr, 0, k_spT, (size_t)NB * DIMC * NSP, (size_t)DIMC * NSP,
      nullptr, DIMC, DIMC);
  // v: tm_sp @ wv -> LIF -> v_spT [t,b,d,n]
  gemm_spike<4, 2, false, false><<<dim3(24, 144), blk, 0, stream>>>(
      tm_sp, aT_c, wv16, 0, sv, bv, nullptr,
      nullptr, 0, v_spT, (size_t)NB * EXPC * NSP, (size_t)EXPC * NSP,
      nullptr, DIMC, EXPC);
  // q: sr_sp @ wq -> float shortcut + broadcast 4-step LIF -> q_sp [t,b,n,c]
  gemm_spike<1, 3, false, false><<<dim3(6, 144), blk, 0, stream>>>(
      sr_sp, 0, wq16, 0, sq, bq, nullptr,
      q_sp, aT_c, nullptr, 0, 0, qsf, DIMC, DIMC);

  kv_kernel<<<dim3(3, 1, TSTEPS * NB * NHEAD), dim3(128), 0, stream>>>(
      k_spT, v_spT, kvT);

  attn_kernel<1><<<dim3(6, 9, NB * NHEAD), blk, 0, stream>>>(
      q_sp, kvT, qq_sp, nullptr);

  const unsigned gDW = (unsigned)((M * EXPC + 255) / 256);         // 55296
  dw_lif_kernel<<<gDW, 256, 0, stream>>>(qq_sp, wdw, sdw, bdw, dw_sp);

  // pw (per-t weights + per-t BN) + shortcut + LIF -> pw_sp
  gemm_spike<4, 2, true, true><<<dim3(6, 144), blk, 0, stream>>>(
      dw_sp, aT_e, wpw16, (size_t)DIMC * EXPC, spw, bpw, qsf,
      pw_sp, aT_c, nullptr, 0, 0, nullptr, EXPC, DIMC);
  // fuse + LIF -> fu_sp
  gemm_spike<4, 2, false, false><<<dim3(6, 144), blk, 0, stream>>>(
      pw_sp, aT_c, wfu16, 0, sfuse, bfuse, nullptr,
      fu_sp, aT_c, nullptr, 0, 0, nullptr, DIMC, DIMC);

  attn_kernel<2><<<dim3(6, 9, NB * NHEAD), blk, 0, stream>>>(
      fu_sp, kvT, x_sp, pooled);

  gate_logits_kernel<<<(TSTEPS * NB * EXPC + 255) / 256, 256, 0, stream>>>(
      pooled, wgate, gl);
  softmax_t_kernel<<<(NB * EXPC + 255) / 256, 256, 0, stream>>>(gl);
  gate_combine_kernel<<<gDW, 256, 0, stream>>>(x_sp, gl, gatesp);

  // proj -> retr (float)
  gemm_spike<1, 0, false, false><<<dim3(6, 144), blk, 0, stream>>>(
      gatesp, 0, wpj16, 0, sproj, bproj, nullptr,
      nullptr, 0, nullptr, 0, 0, retr, EXPC, DIMC);

  s2_kernel<<<gE, 256, 0, stream>>>(search, retr, s2f, s2sp);

  // fc1 -> spike
  gemm_spike<1, 1, false, false><<<dim3(24, 144), blk, 0, stream>>>(
      s2sp, 0, wf116, 0, sfc1, bfc1, nullptr,
      h_sp, 0, nullptr, 0, 0, nullptr, DIMC, EXPC);
  // fc2 -> float
  gemm_spike<1, 0, false, false><<<dim3(6, 144), blk, 0, stream>>>(
      h_sp, 0, wf216, 0, sfc2, bfc2, nullptr,
      nullptr, 0, nullptr, 0, 0, yf, EXPC, DIMC);

  final_kernel<<<gE, 256, 0, stream>>>(s2f, yf, (float*)d_out);
}
